// SelfAttention_18116172055154
// MI455X (gfx1250) — compile-verified, matrix operand order checked by reading, not run on hardware
//
#include <hip/hip_runtime.h>
#include <hip/hip_bf16.h>

typedef __attribute__((ext_vector_type(16))) _Float16 v16h;
typedef __attribute__((ext_vector_type(8)))  _Float16 v8h;
typedef __attribute__((ext_vector_type(8)))  float    v8f;

#define WMMA_F16F32(A, B, C) \
  __builtin_amdgcn_wmma_f32_16x16x32_f16(false, (A), false, (B), (short)0, (C), false, false)

// Async global->LDS copy of 16 bytes (ASYNCcnt-tracked, no VGPR round trip).
// LDS byte offset derived from the destination pointer via addrspacecast to
// LDS (AS3 pointer value == LDS byte offset on AMDGPU device code; cast goes
// through a 64-bit integer so the host semantic pass also accepts it).
// Passing the ptrtoint into the asm makes the shared array ESCAPE, so the
// "memory" clobber correctly orders the later ds_load reads.
__device__ __forceinline__ void async_b128(const void* ldsDst, const void* gsrc) {
  unsigned ldsOff = (unsigned)(unsigned long long)
      (const __attribute__((address_space(3))) void*)ldsDst;
  asm volatile("global_load_async_to_lds_b128 %0, %1, off"
               :: "v"(ldsOff), "v"((unsigned long long)(uintptr_t)gsrc)
               : "memory");
}
__device__ __forceinline__ void wait_async0() {
  asm volatile("s_wait_asynccnt 0x0" ::: "memory");
}

static constexpr int B_   = 16;
static constexpr int N_   = 721;    // tokens (576 patches + 1 cls + 144 scale-1)
static constexpr int NPAD = 736;    // 46 * 16
static constexpr int D_   = 768;
static constexpr int H_   = 12;
static constexpr int HD   = 64;
static constexpr int M_   = B_ * N_;   // 11536 = 721 * 16
static constexpr float INV_SCALE = 0.125f;  // 1/sqrt(64)

// ---------------------------------------------------------------------------
// Utility kernels
// ---------------------------------------------------------------------------
__global__ void k_zero_h(_Float16* __restrict__ p, size_t n) {
  size_t i = (size_t)blockIdx.x * blockDim.x + threadIdx.x;
  size_t stride = (size_t)gridDim.x * blockDim.x;
  for (; i < n; i += stride) p[i] = (_Float16)0.0f;
}

__global__ void k_cvt_h(const float* __restrict__ in, _Float16* __restrict__ out, size_t n) {
  size_t i = (size_t)blockIdx.x * blockDim.x + threadIdx.x;
  size_t stride = (size_t)gridDim.x * blockDim.x;
  for (; i < n; i += stride) out[i] = (_Float16)in[i];
}

// in: [768][768] f32 row-major, out[n*768+k] = in[k*768+n] as f16
__global__ void k_transpose_h(const float* __restrict__ in, _Float16* __restrict__ out) {
  int idx = blockIdx.x * 256 + threadIdx.x;
  int n = idx / D_;
  int k = idx - n * D_;
  out[idx] = (_Float16)in[(size_t)k * D_ + n];
}

// ---------------------------------------------------------------------------
// Fused QKV projection GEMM.
//   grid = (721, 9), block = 256 (8 waves); wg = 16 rows x 256 of 2304 cols,
//   each wave owns a 16x32 tile (two accumulators, A-frag reused).
// ---------------------------------------------------------------------------
__global__ __launch_bounds__(256) void k_qkv_gemm(
    const _Float16* __restrict__ xh,
    const _Float16* __restrict__ wtq, const _Float16* __restrict__ wtk,
    const _Float16* __restrict__ wtv,
    const float* __restrict__ bq, const float* __restrict__ bk,
    const float* __restrict__ bv,
    _Float16* __restrict__ Qh, _Float16* __restrict__ Kh, _Float16* __restrict__ Vt) {
  __shared__ alignas(32) _Float16 aTile[16 * D_];   // 24 KB A row-block

  const int tid  = threadIdx.x;
  const int row0 = blockIdx.x * 16;

  // Async stage of the 16-row A block (contiguous 24 KB) into LDS.
  #pragma unroll
  for (int i = 0; i < (16 * D_) / (8 * 256); ++i) {
    int idx = tid + i * 256;                       // 16-byte chunk index
    async_b128(&aTile[idx * 8], &xh[(size_t)row0 * D_ + (size_t)idx * 8]);
  }
  wait_async0();
  __syncthreads();

  const int wave = tid >> 5;
  const int lane = tid & 31;
  const int lr   = lane & 15;
  const int hl   = lane >> 4;

  const int cg0   = blockIdx.y * 256 + wave * 32;   // 0..2303 (never crosses q/k/v)
  const int which = cg0 / D_;                       // 0=q 1=k 2=v
  const int cc0   = cg0 - which * D_;

  const _Float16* wt = (which == 0) ? wtq : (which == 1) ? wtk : wtv;
  const float* bias  = (which == 0) ? bq  : (which == 1) ? bk  : bv;
  const _Float16* w0row = wt + (size_t)(cc0 + lr) * D_;
  const _Float16* w1row = wt + (size_t)(cc0 + 16 + lr) * D_;

  v8f acc[2] = {{}, {}};
  #pragma unroll 4
  for (int k = 0; k < D_; k += 32) {
    v16h a  = *reinterpret_cast<const v16h*>(&aTile[lr * D_ + k + hl * 16]);
    v16h b0 = *reinterpret_cast<const v16h*>(&w0row[k + hl * 16]);
    v16h b1 = *reinterpret_cast<const v16h*>(&w1row[k + hl * 16]);
    __builtin_prefetch(&w0row[k + 256], 0, 0);
    acc[0] = WMMA_F16F32(a, b0, acc[0]);
    acc[1] = WMMA_F16F32(a, b1, acc[1]);
  }

  #pragma unroll
  for (int ct = 0; ct < 2; ++ct) {
    const int col = cc0 + ct * 16 + lr;
    const int h   = col >> 6;
    const int j   = col & 63;
    const float bb = bias[col];
    #pragma unroll
    for (int r = 0; r < 8; ++r) {
      int g  = row0 + r + hl * 8;
      int b_ = g / N_;
      int t  = g - b_ * N_;
      _Float16 val = (_Float16)(acc[ct][r] + bb);
      size_t bhh = (size_t)(b_ * H_ + h);
      if (which == 0)      Qh[(bhh * NPAD + t) * HD + j] = val;
      else if (which == 1) Kh[(bhh * NPAD + t) * HD + j] = val;
      else                 Vt[(bhh * HD + j) * NPAD + t] = val;
    }
  }
}

// ---------------------------------------------------------------------------
// Flash attention with fused RW column-weighted softmax.
//   grid = (6, 12, 16), block = 256 (8 waves); wave owns 16 query rows.
//   K/V^T slabs for each 32-key step are async-staged in LDS, shared by all
//   8 waves (8x global-traffic reduction). Tail waves clamp rows, mask stores.
// ---------------------------------------------------------------------------
__global__ __launch_bounds__(256) void k_attn(
    const _Float16* __restrict__ Qh, const _Float16* __restrict__ Kh,
    const _Float16* __restrict__ Vt, _Float16* __restrict__ outh) {
  // [0,2048)     kTile: 32 keys x 64    (4 KB)
  // [2048,4096)  vTile: 64 j x 32 keys  (4 KB)
  // [4096,8192)  plds : 8 waves x 16x32 (8 KB)
  __shared__ alignas(32) _Float16 smem[8192];
  constexpr int KOFF = 0, VOFF = 2048, POFF = 4096;

  const int tid  = threadIdx.x;
  const int wave = tid >> 5;
  const int lane = tid & 31;
  const int lr   = lane & 15;
  const int hl   = lane >> 4;

  const int b = blockIdx.z, h = blockIdx.y;
  int row0 = blockIdx.x * 128 + wave * 16;
  const bool active = (row0 < NPAD);
  if (!active) row0 = NPAD - 16;        // clamp: keep wave in lockstep for barriers

  const size_t bh = (size_t)(b * H_ + h);
  const _Float16* q  = Qh + bh * NPAD * HD;
  const _Float16* kk = Kh + bh * NPAD * HD;
  const _Float16* vt = Vt + bh * HD * NPAD;

  // Resident Q A-fragments (hd = 64 -> two K=32 chunks).
  v16h qa0 = *reinterpret_cast<const v16h*>(&q[(row0 + lr) * HD + hl * 16]);
  v16h qa1 = *reinterpret_cast<const v16h*>(&q[(row0 + lr) * HD + 32 + hl * 16]);

  float m[8], d[8];
  #pragma unroll
  for (int r = 0; r < 8; ++r) { m[r] = -3.0e38f; d[r] = 0.0f; }
  v8f O0 = {}, O1 = {}, O2 = {}, O3 = {};

  for (int c0 = 0; c0 < NPAD; c0 += 32) {
    // ---- Async cooperative stage: K slab 32x64, V^T slab 64x32 ----
    {
      // K: 4 KB = 256 x 16B chunks, one per thread
      async_b128(&smem[KOFF + (tid >> 3) * 64 + (tid & 7) * 8],
                 &kk[(size_t)(c0 + (tid >> 3)) * HD + (tid & 7) * 8]);
      // V^T: 4 KB = 256 x 16B chunks, one per thread
      async_b128(&smem[VOFF + (tid >> 2) * 32 + (tid & 3) * 8],
                 &vt[(size_t)(tid >> 2) * NPAD + c0 + (tid & 3) * 8]);
      wait_async0();
    }
    __syncthreads();

    // ---- S tiles: two 16x16 f32 C-frags over keys [c0, c0+32) ----
    v8f S0 = {}, S1 = {};
    {
      v16h b0 = *reinterpret_cast<const v16h*>(&smem[KOFF + lr * 64 + hl * 16]);
      S0 = WMMA_F16F32(qa0, b0, S0);
      v16h b1 = *reinterpret_cast<const v16h*>(&smem[KOFF + lr * 64 + 32 + hl * 16]);
      S0 = WMMA_F16F32(qa1, b1, S0);
      v16h b2 = *reinterpret_cast<const v16h*>(&smem[KOFF + (16 + lr) * 64 + hl * 16]);
      S1 = WMMA_F16F32(qa0, b2, S1);
      v16h b3 = *reinterpret_cast<const v16h*>(&smem[KOFF + (16 + lr) * 64 + 32 + hl * 16]);
      S1 = WMMA_F16F32(qa1, b3, S1);
    }

    const int col0 = c0 + lr;
    const int col1 = c0 + 16 + lr;
    const float w0 = (col0 >= N_) ? 0.0f : ((col0 >= 577) ? 0.5f : 1.0f);
    const float w1 = (col1 >= N_) ? 0.0f : ((col1 >= 577) ? 0.5f : 1.0f);

    #pragma unroll
    for (int r = 0; r < 8; ++r) {
      float s0 = (col0 < N_) ? S0[r] * INV_SCALE : -1.0e30f;
      float s1 = (col1 < N_) ? S1[r] * INV_SCALE : -1.0e30f;
      float mx = fmaxf(s0, s1);
      #pragma unroll
      for (int off = 1; off < 16; off <<= 1) mx = fmaxf(mx, __shfl_xor(mx, off, 32));
      float nm    = fmaxf(m[r], mx);
      float alpha = __expf(m[r] - nm);
      m[r] = nm;
      float p0 = __expf(s0 - nm) * w0;
      float p1 = __expf(s1 - nm) * w1;
      float ps = p0 + p1;
      #pragma unroll
      for (int off = 1; off < 16; off <<= 1) ps += __shfl_xor(ps, off, 32);
      d[r] = d[r] * alpha + ps;
      O0[r] *= alpha; O1[r] *= alpha; O2[r] *= alpha; O3[r] *= alpha;
      // stage P (C-layout) into per-wave LDS row-major 16x32
      smem[POFF + wave * 512 + (r + hl * 8) * 32 + lr]      = (_Float16)p0;
      smem[POFF + wave * 512 + (r + hl * 8) * 32 + 16 + lr] = (_Float16)p1;
    }

    // Reload P as A-fragment (same-wave LDS dependency -> DScnt wait by backend).
    v16h pa = *reinterpret_cast<const v16h*>(&smem[POFF + wave * 512 + lr * 32 + hl * 16]);

    // ---- O += P * V : B-frags from LDS V^T slab ----
    v16h vb0 = *reinterpret_cast<const v16h*>(&smem[VOFF + (0 * 16 + lr) * 32 + hl * 16]);
    O0 = WMMA_F16F32(pa, vb0, O0);
    v16h vb1 = *reinterpret_cast<const v16h*>(&smem[VOFF + (1 * 16 + lr) * 32 + hl * 16]);
    O1 = WMMA_F16F32(pa, vb1, O1);
    v16h vb2 = *reinterpret_cast<const v16h*>(&smem[VOFF + (2 * 16 + lr) * 32 + hl * 16]);
    O2 = WMMA_F16F32(pa, vb2, O2);
    v16h vb3 = *reinterpret_cast<const v16h*>(&smem[VOFF + (3 * 16 + lr) * 32 + hl * 16]);
    O3 = WMMA_F16F32(pa, vb3, O3);

    __syncthreads();   // protect K/V slabs before next iteration's async overwrite
  }

  // Finalize: divide by weighted denominator, store f16 [b*721+t][h*64+j].
  if (active) {
    #pragma unroll
    for (int r = 0; r < 8; ++r) {
      int t = row0 + r + hl * 8;
      if (t < N_) {
        float inv = 1.0f / d[r];
        size_t base = ((size_t)(b * N_ + t)) * D_ + h * HD;
        outh[base + 0  + lr] = (_Float16)(O0[r] * inv);
        outh[base + 16 + lr] = (_Float16)(O1[r] * inv);
        outh[base + 32 + lr] = (_Float16)(O2[r] * inv);
        outh[base + 48 + lr] = (_Float16)(O3[r] * inv);
      }
    }
  }
}

// ---------------------------------------------------------------------------
// Output projection: out[11536 x 768] f32 = attn_out(f16) x Wo + bo
//   grid = (721, 3), block = 256 (8 waves, 16 rows x 256 cols per wg)
// ---------------------------------------------------------------------------
__global__ __launch_bounds__(256) void k_out_gemm(
    const _Float16* __restrict__ ah, const _Float16* __restrict__ wot,
    const float* __restrict__ bo, float* __restrict__ out) {
  __shared__ alignas(32) _Float16 aTile[16 * D_];

  const int tid  = threadIdx.x;
  const int row0 = blockIdx.x * 16;

  #pragma unroll
  for (int i = 0; i < (16 * D_) / (8 * 256); ++i) {
    int idx = tid + i * 256;
    async_b128(&aTile[idx * 8], &ah[(size_t)row0 * D_ + (size_t)idx * 8]);
  }
  wait_async0();
  __syncthreads();

  const int wave = tid >> 5;
  const int lane = tid & 31;
  const int lr   = lane & 15;
  const int hl   = lane >> 4;
  const int cb0  = blockIdx.y * 256 + wave * 32;

  const _Float16* w0row = wot + (size_t)(cb0 + lr) * D_;
  const _Float16* w1row = wot + (size_t)(cb0 + 16 + lr) * D_;

  v8f acc[2] = {{}, {}};
  #pragma unroll 4
  for (int k = 0; k < D_; k += 32) {
    v16h a  = *reinterpret_cast<const v16h*>(&aTile[lr * D_ + k + hl * 16]);
    v16h b0 = *reinterpret_cast<const v16h*>(&w0row[k + hl * 16]);
    v16h b1 = *reinterpret_cast<const v16h*>(&w1row[k + hl * 16]);
    __builtin_prefetch(&w0row[k + 256], 0, 0);
    acc[0] = WMMA_F16F32(a, b0, acc[0]);
    acc[1] = WMMA_F16F32(a, b1, acc[1]);
  }

  #pragma unroll
  for (int ct = 0; ct < 2; ++ct) {
    const int col = cb0 + ct * 16 + lr;
    const float bb = bo[col];
    #pragma unroll
    for (int r = 0; r < 8; ++r) {
      out[(size_t)(row0 + r + hl * 8) * D_ + col] = acc[ct][r] + bb;
    }
  }
}

// ---------------------------------------------------------------------------
extern "C" void kernel_launch(void* const* d_in, const int* in_sizes, int n_in,
                              void* d_out, int out_size, void* d_ws, size_t ws_size,
                              hipStream_t stream) {
  const float* x  = (const float*)d_in[0];
  const float* Wq = (const float*)d_in[1];
  const float* bq = (const float*)d_in[2];
  const float* Wk = (const float*)d_in[3];
  const float* bk = (const float*)d_in[4];
  const float* Wv = (const float*)d_in[5];
  const float* bv = (const float*)d_in[6];
  const float* Wo = (const float*)d_in[7];
  const float* bo = (const float*)d_in[8];
  float* out = (float*)d_out;

  char* ws = (char*)d_ws;
  size_t off = 0;
  auto take = [&](size_t halves) {
    char* p = ws + off;
    off += halves * sizeof(_Float16);
    return (_Float16*)p;
  };
  _Float16* xh  = take((size_t)M_ * D_);
  _Float16* wtq = take((size_t)D_ * D_);
  _Float16* wtk = take((size_t)D_ * D_);
  _Float16* wtv = take((size_t)D_ * D_);
  _Float16* wot = take((size_t)D_ * D_);
  _Float16* Qh  = take((size_t)B_ * H_ * NPAD * HD);
  _Float16* Kh  = take((size_t)B_ * H_ * NPAD * HD);
  _Float16* Vt  = take((size_t)B_ * H_ * NPAD * HD);
  _Float16* oh  = take((size_t)M_ * D_);
  (void)ws_size; (void)in_sizes; (void)n_in; (void)out_size;

  // Zero padded Q/K/V (contiguous) so pad lanes never feed NaN into WMMA.
  k_zero_h<<<4096, 256, 0, stream>>>(Qh, (size_t)B_ * H_ * NPAD * HD * 3);

  // Pack inputs to f16 (weights transposed to [N][K] for contiguous B-frags).
  k_cvt_h<<<8192, 256, 0, stream>>>(x, xh, (size_t)M_ * D_);
  k_transpose_h<<<D_ * D_ / 256, 256, 0, stream>>>(Wq, wtq);
  k_transpose_h<<<D_ * D_ / 256, 256, 0, stream>>>(Wk, wtk);
  k_transpose_h<<<D_ * D_ / 256, 256, 0, stream>>>(Wv, wtv);
  k_transpose_h<<<D_ * D_ / 256, 256, 0, stream>>>(Wo, wot);

  // Fused QKV projection (2304 output columns across q/k/v).
  k_qkv_gemm<<<dim3(M_ / 16, 2304 / 256), 256, 0, stream>>>(
      xh, wtq, wtk, wtv, bq, bk, bv, Qh, Kh, Vt);

  // Flash attention with fused RW column-weighted softmax.
  k_attn<<<dim3(6, H_, B_), 256, 0, stream>>>(Qh, Kh, Vt, oh);

  // Output projection + bias -> f32.
  k_out_gemm<<<dim3(M_ / 16, D_ / 256), 256, 0, stream>>>(oh, wot, bo, out);
}
